// ArcFaceLoss_40939628265631
// MI455X (gfx1250) — matrix operand order repaired and checked by reading, not compiled, not run on hardware
//
#include <hip/hip_runtime.h>
#include <hip/hip_bf16.h>

// ---------- types matching the gfx1250 builtin signatures ----------
typedef __attribute__((ext_vector_type(16))) __bf16        v16bf;
typedef __attribute__((ext_vector_type(8)))  float         v8f;
typedef __attribute__((ext_vector_type(4)))  float         f32x4;
typedef __attribute__((ext_vector_type(4)))  unsigned int  v4u;
typedef __attribute__((ext_vector_type(8)))  int           v8i;
typedef __attribute__((ext_vector_type(4)))  int           v4i;

#define BATCH_N   1024
#define EMB_D     512
#define NCLS      100000
#define SCALE_S   64.0f
#define MARGIN_M  0.5f

union FragBF {
  v16bf v;
  f32x4 f[2];
};

// ============================================================================
// Kernel 1: normalize embedding rows; write f32 + bf16 hi/lo split; zero sums.
// ============================================================================
__global__ __launch_bounds__(256) void af_prep_embed(
    const float* __restrict__ e, float* __restrict__ e_norm,
    __bf16* __restrict__ e_hi, __bf16* __restrict__ e_lo,
    float* __restrict__ sums)
{
  const int wave = threadIdx.x >> 5;
  const int lane = threadIdx.x & 31;
  const int row  = blockIdx.x * 8 + wave;
  const float* er = e + (size_t)row * EMB_D;

  float v[16];
  float sq = 0.f;
#pragma unroll
  for (int i = 0; i < 16; ++i) {
    v[i] = er[lane + i * 32];
    sq += v[i] * v[i];
  }
#pragma unroll
  for (int m = 16; m >= 1; m >>= 1) sq += __shfl_xor(sq, m, 32);
  const float rn = 1.f / fmaxf(sqrtf(sq), 1e-12f);

#pragma unroll
  for (int i = 0; i < 16; ++i) {
    const int k = lane + i * 32;
    const float en = v[i] * rn;
    e_norm[(size_t)row * EMB_D + k] = en;
    const __bf16 h = (__bf16)en;
    e_hi[(size_t)row * EMB_D + k] = h;
    e_lo[(size_t)row * EMB_D + k] = (__bf16)(en - (float)h);
  }
  if (lane == 0) sums[row] = 0.f;
}

// ============================================================================
// Kernel 2: per-row target column (exact f32): cos and cos(acos(clip)+m).
// ============================================================================
__global__ __launch_bounds__(256) void af_target(
    const float* __restrict__ w, const int* __restrict__ labels,
    const float* __restrict__ e_norm, float* __restrict__ tcos,
    float* __restrict__ tcosm)
{
  const int wave = threadIdx.x >> 5;
  const int lane = threadIdx.x & 31;
  const int row  = blockIdx.x * 8 + wave;
  const int lbl  = labels[row];
  const float* wr = w + (size_t)lbl * EMB_D;
  const float* er = e_norm + (size_t)row * EMB_D;

  float dot = 0.f, sq = 0.f;
#pragma unroll
  for (int i = 0; i < 16; ++i) {
    const int k = lane + i * 32;
    const float wv = wr[k];
    dot += er[k] * wv;
    sq  += wv * wv;
  }
#pragma unroll
  for (int m = 16; m >= 1; m >>= 1) {
    dot += __shfl_xor(dot, m, 32);
    sq  += __shfl_xor(sq,  m, 32);
  }
  if (lane == 0) {
    const float rn = 1.f / fmaxf(sqrtf(sq), 1e-12f);
    const float c  = dot * rn;
    const float cc = fminf(fmaxf(c, -1.f + 1e-7f), 1.f - 1e-7f);
    const float cm = cosf(acosf(cc) + MARGIN_M);
    tcos[row]  = c;
    tcosm[row] = cm;
  }
}

// ============================================================================
// Kernel 3: fused GEMM + streaming sum-exp with TDM double-buffered W tiles.
//  - Wave 0 issues TENSOR_LOAD_TO_LDS for the f32 16x32 W tile of step k+1
//    while all waves run WMMA on the tile of step k (TENSORcnt-tracked).
//  - Each wave reads the whole f32 tile for its own B fragments, doing the
//    bf16 hi/lo split in registers; per-class ||w||^2 falls out for free
//    (per-lane accumulate + one shfl_xor(16)) -- no LDS reduction needed.
//  - 3-term split product: hi*hi + lo*hi + hi*lo (~fp32 accuracy).
//  - Epilogue: exp(64*cos - 64), 16-lane class reduction, atomicAdd per row.
// ============================================================================
__global__ __launch_bounds__(256) void af_gemm_lse(
    const float* __restrict__ w, const __bf16* __restrict__ e_hi,
    const __bf16* __restrict__ e_lo, float* __restrict__ sums)
{
  __shared__ float ldsW[2][16 * 32];   // double-buffered f32 W tile, 2 x 2KB

  const int t    = threadIdx.x;
  const int wave = t >> 5;
  const int lane = t & 31;
  const int half = lane >> 4;     // selects K half: k = 16*half + s (contiguous order)
  const int n16  = lane & 15;     // class within tile (B/C/D lane dim)
  const int cls0 = blockIdx.x * 16;

  const unsigned long long gbase =
      (unsigned long long)(uintptr_t)(w + (size_t)cls0 * EMB_D);
  const unsigned int ldsA = (unsigned int)(uintptr_t)&ldsW[0][0];
  const unsigned int ldsB = (unsigned int)(uintptr_t)&ldsW[1][0];

  // ---- TDM D# issue: 2-D tensor, 4-byte elems, tile 32(k) x 16(classes) ----
  auto tdm_issue = [&](int k0, unsigned int ldsAddr) {
    const unsigned long long ga = gbase + (unsigned long long)k0 * 4ull;
    v4u g0;
    g0[0] = 1u;                                          // count=1 (valid D#)
    g0[1] = ldsAddr;                                     // lds_addr (bytes)
    g0[2] = (unsigned int)(ga & 0xFFFFFFFFull);          // global_addr[31:0]
    g0[3] = (unsigned int)((ga >> 32) & 0x1FFFFFFu)      // global_addr[56:32]
          | (2u << 30);                                  // type=2 ("image")
    v8i g1;
    g1[0] = (int)(2u << 16);        // workgroup_mask=0, data_size=2 -> 4 bytes
    g1[1] = (int)(512u << 16);      // tensor_dim0 = 512   (bits 79:48, lo16)
    g1[2] = (int)(16u << 16);       // dim0 hi16=0 | tensor_dim1=16 lo16
    g1[3] = (int)(32u << 16);       // dim1 hi16=0 | tile_dim0=32
    g1[4] = (int)(16u);             // tile_dim1=16, tile_dim2=0
    g1[5] = (int)(512u);            // tensor_dim0_stride = 512 (lo32)
    g1[6] = 0;                      // stride hi16 | dim1_stride lo16 (unused 2D)
    g1[7] = 0;
    const v4i z4 = {0, 0, 0, 0};                 // groups 2/3 unused (2-D)
    const v8i z8 = {0, 0, 0, 0, 0, 0, 0, 0};     // extra group (6-arg form)
    __builtin_amdgcn_tensor_load_to_lds(g0, g1, z4, z4, z8, 0);
  };

  v8f acc[8] = {};                  // 8 M-tiles of 16 rows -> 128 rows/wave
  float sqp = 0.f;                  // per-lane partial of ||w_class||^2

  if (t < 32) tdm_issue(0, ldsA);   // prologue: buf0 <- k-step 0

  const int NK = EMB_D / 32;        // 16 k-steps
  for (int ks = 0; ks < NK; ++ks) {
    const int buf = ks & 1;
    __syncthreads();                // readers of buf^1 (step ks-1) are done
    if (t < 32) {
      if (ks + 1 < NK) {
        tdm_issue((ks + 1) * 32, buf ? ldsA : ldsB);   // prefetch next tile
        __builtin_amdgcn_s_wait_tensorcnt(1);          // tile ks complete
      } else {
        __builtin_amdgcn_s_wait_tensorcnt(0);
      }
    }
    __syncthreads();                // publish buf ks to all waves

    // ---- B fragments: 16 contiguous f32 per lane; split to bf16 hi/lo ----
    const float* bsrc = &ldsW[buf][n16 * 32 + half * 16];
    f32x4 bq[4];
#pragma unroll
    for (int q = 0; q < 4; ++q) bq[q] = ((const f32x4*)bsrc)[q];
    FragBF bh, bl;
#pragma unroll
    for (int q = 0; q < 4; ++q) {
#pragma unroll
      for (int j = 0; j < 4; ++j) {
        const float f = bq[q][j];
        sqp += f * f;
        const __bf16 hb = (__bf16)f;
        bh.v[q * 4 + j] = hb;
        bl.v[q * 4 + j] = (__bf16)(f - (float)hb);
      }
    }

    const int koff = ks * 32 + half * 16;  // A layout: K=h+half*16
#pragma unroll
    for (int mt = 0; mt < 8; ++mt) {
      const int m = wave * 128 + mt * 16 + n16;        // A lane dim = M
      FragBF ah, al;
      const f32x4* pah = (const f32x4*)(e_hi + (size_t)m * EMB_D + koff);
      const f32x4* pal = (const f32x4*)(e_lo + (size_t)m * EMB_D + koff);
      ah.f[0] = pah[0]; ah.f[1] = pah[1];              // 16 contiguous halves
      al.f[0] = pal[0]; al.f[1] = pal[1];

      acc[mt] = __builtin_amdgcn_wmma_f32_16x16x32_bf16(
          false, ah.v, false, bh.v, (short)0, acc[mt], false, false);
      acc[mt] = __builtin_amdgcn_wmma_f32_16x16x32_bf16(
          false, al.v, false, bh.v, (short)0, acc[mt], false, false);
      acc[mt] = __builtin_amdgcn_wmma_f32_16x16x32_bf16(
          false, ah.v, false, bl.v, (short)0, acc[mt], false, false);
    }
  }

  // ---- per-class inverse norm: combine the two K-halves of this class ----
  const float sqc = sqp + __shfl_xor(sqp, 16, 32);
  const float rnc = 1.f / fmaxf(sqrtf(sqc), 1e-12f);

  // ---- epilogue: fixed-shift exp, reduce over 16 classes, atomic per row ----
#pragma unroll
  for (int mt = 0; mt < 8; ++mt) {
#pragma unroll
    for (int r = 0; r < 8; ++r) {
      // D layout: value at (M = wave*128 + mt*16 + r + half*8, N = n16)
      float p = __expf(fmaf(SCALE_S * rnc, acc[mt][r], -SCALE_S));
#pragma unroll
      for (int msk = 1; msk < 16; msk <<= 1) p += __shfl_xor(p, msk, 16);
      if (n16 == 0)
        atomicAdd(&sums[wave * 128 + mt * 16 + r + half * 8], p);
    }
  }
}

// ============================================================================
// Kernel 4: swap label column's exp for the margin term, finish LSE, mean.
// ============================================================================
__global__ __launch_bounds__(256) void af_finalize(
    const float* __restrict__ sums, const float* __restrict__ tcos,
    const float* __restrict__ tcosm, float* __restrict__ out)
{
  __shared__ float red[256];
  const int t = threadIdx.x;
  float local = 0.f;
#pragma unroll
  for (int i = 0; i < 4; ++i) {
    const int b = t + i * 256;
    const float cm = tcosm[b];
    const float cr = tcos[b];
    float S = sums[b] + __expf(fmaf(SCALE_S, cm, -SCALE_S))
                      - __expf(fmaf(SCALE_S, cr, -SCALE_S));
    S = fmaxf(S, 1e-38f);
    local += SCALE_S + __logf(S) - SCALE_S * cm;   // lse - 64*cos_m
  }
  red[t] = local;
  __syncthreads();
  for (int s = 128; s > 0; s >>= 1) {
    if (t < s) red[t] += red[t + s];
    __syncthreads();
  }
  if (t == 0) out[0] = red[0] * (1.f / (float)BATCH_N);
}

// ============================================================================
extern "C" void kernel_launch(void* const* d_in, const int* in_sizes, int n_in,
                              void* d_out, int out_size, void* d_ws, size_t ws_size,
                              hipStream_t stream) {
  (void)in_sizes; (void)n_in; (void)out_size; (void)ws_size;
  const float* emb    = (const float*)d_in[0];
  const float* wgt    = (const float*)d_in[1];
  const int*   labels = (const int*)d_in[2];
  float* out = (float*)d_out;

  char* ws = (char*)d_ws;
  float*  e_norm = (float*)(ws);                         // 2 MB
  __bf16* e_hi   = (__bf16*)(ws + (2u << 20));           // 1 MB
  __bf16* e_lo   = (__bf16*)(ws + (3u << 20));           // 1 MB
  float*  sums   = (float*)(ws + (4u << 20));            // 4 KB
  float*  tcos   = (float*)(ws + (4u << 20) + 4096);     // 4 KB
  float*  tcosm  = (float*)(ws + (4u << 20) + 8192);     // 4 KB

  af_prep_embed<<<BATCH_N / 8, 256, 0, stream>>>(emb, e_norm, e_hi, e_lo, sums);
  af_target    <<<BATCH_N / 8, 256, 0, stream>>>(wgt, labels, e_norm, tcos, tcosm);
  af_gemm_lse  <<<NCLS / 16,   256, 0, stream>>>(wgt, e_hi, e_lo, sums);
  af_finalize  <<<1,           256, 0, stream>>>(sums, tcos, tcosm, out);
}
